// GAT2_23304492548679
// MI455X (gfx1250) — matrix operand order repaired and checked by reading, not compiled, not run on hardware
//
#include <hip/hip_runtime.h>
#include <hip/hip_bf16.h>

#define N_NODES   10000
#define N_EDGES   160000
#define N_ETOT    (N_EDGES + N_NODES)   // with self loops
#define VOCAB     1024
#define C1        512
#define C2        128
#define N_CLASSES 16
#define N_GRAPHS  64
#define NEG_SLOPE 0.2f

typedef unsigned short u16x8  __attribute__((ext_vector_type(8)));
typedef unsigned short u16x16 __attribute__((ext_vector_type(16)));
typedef __bf16         v16bf  __attribute__((ext_vector_type(16)));
typedef float          v8f    __attribute__((ext_vector_type(8)));

// ---------- helpers ----------
__device__ __forceinline__ unsigned short f2bf(float f) {
  unsigned int u = __float_as_uint(f);
  unsigned int r = (u + 0x7FFFu + ((u >> 16) & 1u)) >> 16;   // RNE
  return (unsigned short)r;
}

__device__ __forceinline__ v16bf load_a_frag(const unsigned short* p) {
  u16x8 x0 = *(const u16x8*)(p);
  u16x8 x1 = *(const u16x8*)(p + 16);
  u16x16 u = __builtin_shufflevector(x0, x1, 0,1,2,3,4,5,6,7,8,9,10,11,12,13,14,15);
  return __builtin_bit_cast(v16bf, u);
}

// ---------- f32 -> bf16 ----------
__global__ void k_cvt_bf16(const float* __restrict__ in, unsigned short* __restrict__ out, int n) {
  int i = blockIdx.x * blockDim.x + threadIdx.x;
  if (i < n) out[i] = f2bf(in[i]);
}

// ---------- pack W[K][N] (f32 row-major) into WMMA-B bf16 layout ----------
// Wp index = ((nT*KT + kt)*32 + lane)*16 + e ;  col = nT*16 + (lane&15),
// k = kt*32 + (e>>3)*16 + (lane>>4)*8 + (e&7)
__global__ void k_pack_w(const float* __restrict__ W, unsigned short* __restrict__ Wp, int K, int N) {
  int idx = blockIdx.x * blockDim.x + threadIdx.x;
  int total = K * N;
  if (idx >= total) return;
  int e    = idx & 15;
  int lane = (idx >> 4) & 31;
  int g    = idx >> 9;           // (nT*KT + kt)
  int KT   = K >> 5;
  int nT   = g / KT;
  int kt   = g - nT * KT;
  int col  = nT * 16 + (lane & 15);
  int k    = kt * 32 + (e >> 3) * 16 + (lane >> 4) * 8 + (e & 7);
  Wp[idx] = f2bf(W[(size_t)k * N + col]);
}

// ---------- bf16 WMMA GEMM: C[M][N] = A[M][K] * W + bias ----------
// register-blocked: one wave computes a BM x BN grid of 16x16 tiles.
// per k-step: BM A-fragments + BN B-fragments -> BM*BN v_wmma_f32_16x16x32_bf16
#define BM 2
#define BN 4

__global__ void k_gemm_bf16(const unsigned short* __restrict__ A,
                            const unsigned short* __restrict__ Bp,
                            const float* __restrict__ bias,
                            float* __restrict__ Cout,
                            int M, int N, int K) {
  int wave = (blockIdx.x * blockDim.x + threadIdx.x) >> 5;
  int lane = threadIdx.x & 31;
  int tilesM = M >> 4, tilesN = N >> 4;
  int mBlocks = (tilesM + BM - 1) / BM;
  int nBlocks = tilesN / BN;                    // N is a multiple of 64 here
  if (wave >= mBlocks * nBlocks) return;        // uniform per-wave: EXEC stays all-ones
  int mb = wave % mBlocks;
  int nb = wave / mBlocks;
  int half = lane >> 4, lr = lane & 15;
  int KT = K >> 5;
  int mT0 = mb * BM;
  bool m1ok = (mT0 + 1) < tilesM;               // uniform remainder guard (tilesM may be odd)

  v8f c[BM][BN];
  #pragma unroll
  for (int i = 0; i < BM; ++i)
    #pragma unroll
    for (int j = 0; j < BN; ++j)
      c[i][j] = (v8f){0.f, 0.f, 0.f, 0.f, 0.f, 0.f, 0.f, 0.f};

  const unsigned short* arow0 = A + (size_t)(mT0 * 16 + lr) * K + half * 8;
  const unsigned short* arow1 = m1ok ? (arow0 + (size_t)16 * K) : arow0;  // safe duplicate
  const unsigned short* bp[BN];
  #pragma unroll
  for (int j = 0; j < BN; ++j)
    bp[j] = Bp + ((size_t)((nb * BN + j) * KT) * 32 + lane) * 16;

  for (int kt = 0; kt < KT; ++kt) {
    v16bf a0 = load_a_frag(arow0);
    v16bf a1 = load_a_frag(arow1);
    v16bf bv[BN];
    #pragma unroll
    for (int j = 0; j < BN; ++j)
      bv[j] = __builtin_bit_cast(v16bf, *(const u16x16*)bp[j]);

    #pragma unroll
    for (int j = 0; j < BN; ++j)
      c[0][j] = __builtin_amdgcn_wmma_f32_16x16x32_bf16(false, a0, false, bv[j], (short)0, c[0][j], false, false);
    #pragma unroll
    for (int j = 0; j < BN; ++j)
      c[1][j] = __builtin_amdgcn_wmma_f32_16x16x32_bf16(false, a1, false, bv[j], (short)0, c[1][j], false, false);

    arow0 += 32;
    arow1 += 32;
    #pragma unroll
    for (int j = 0; j < BN; ++j) bp[j] += 32 * 16;
  }

  #pragma unroll
  for (int i = 0; i < BM; ++i) {
    if (i > 0 && !m1ok) break;                  // uniform
    float* orow = Cout + (size_t)((mT0 + i) * 16 + half * 8) * N + (nb * BN) * 16 + lr;
    #pragma unroll
    for (int j = 0; j < BN; ++j) {
      float b = bias[(nb * BN + j) * 16 + lr];
      #pragma unroll
      for (int v = 0; v < 8; ++v)
        orow[(size_t)v * N + j * 16] = c[i][j][v] + b;
    }
  }
}

// ---------- CSR build over destination node ----------
__global__ void k_edge_count(const int* __restrict__ ei, int* __restrict__ deg) {
  int e = blockIdx.x * blockDim.x + threadIdx.x;
  if (e >= N_ETOT) return;
  int dst = (e < N_EDGES) ? ei[N_EDGES + e] : (e - N_EDGES);
  atomicAdd(&deg[dst], 1);
}

__global__ void k_scan(const int* __restrict__ deg, int* __restrict__ rowptr, int n) {
  __shared__ int buf[256];
  __shared__ int carry;
  int tid = threadIdx.x;
  if (tid == 0) carry = 0;
  __syncthreads();
  for (int base = 0; base < n; base += 256) {
    int i = base + tid;
    int v = (i < n) ? deg[i] : 0;
    buf[tid] = v;
    __syncthreads();
    for (int off = 1; off < 256; off <<= 1) {
      int t = (tid >= off) ? buf[tid - off] : 0;
      __syncthreads();
      buf[tid] += t;
      __syncthreads();
    }
    if (i < n) rowptr[i] = carry + buf[tid] - v;   // exclusive
    __syncthreads();
    if (tid == 255) carry += buf[255];
    __syncthreads();
  }
  if (tid == 0) rowptr[n] = carry;
}

__global__ void k_edge_fill(const int* __restrict__ ei, const int* __restrict__ rowptr,
                            int* __restrict__ fill, int* __restrict__ csrsrc) {
  int e = blockIdx.x * blockDim.x + threadIdx.x;
  if (e >= N_ETOT) return;
  int src, dst;
  if (e < N_EDGES) { src = ei[e]; dst = ei[N_EDGES + e]; }
  else             { src = dst = e - N_EDGES; }
  int pos = rowptr[dst] + atomicAdd(&fill[dst], 1);
  csrsrc[pos] = src;
}

// ---------- GATv2 aggregation: one 128-thread block per destination node ----------
#define AGG_T 128
#define SMAX  1024

template <int C>
__device__ __forceinline__ float edge_score(const float* __restrict__ xl, int src,
                                            const float* xr_s, const float* att_s,
                                            float* red, int tid) {
  float p = 0.f;
  #pragma unroll
  for (int q = 0; q < C / AGG_T; ++q) {
    int c = tid + q * AGG_T;
    float v = xl[(size_t)src * C + c] + xr_s[c];
    v = v > 0.f ? v : NEG_SLOPE * v;
    p += v * att_s[c];
  }
  red[tid] = p;
  __syncthreads();
  for (int off = 64; off > 0; off >>= 1) {
    if (tid < off) red[tid] += red[tid + off];
    __syncthreads();
  }
  float e = red[0];
  __syncthreads();
  return e;
}

template <int C>
__global__ void k_gat_agg(const float* __restrict__ xl, const float* __restrict__ xr,
                          const float* __restrict__ att, const float* __restrict__ bias,
                          const int* __restrict__ rowptr, const int* __restrict__ csrsrc,
                          float* __restrict__ out) {
  int node = blockIdx.x;
  int tid  = threadIdx.x;
  __shared__ float xr_s[C];
  __shared__ float att_s[C];
  __shared__ float red[AGG_T];
  __shared__ float sc[SMAX];
  __shared__ float m_s, s_s;

  #pragma unroll
  for (int q = 0; q < C / AGG_T; ++q) {
    int c = tid + q * AGG_T;
    xr_s[c]  = xr[(size_t)node * C + c];
    att_s[c] = att[c];
  }
  if (tid == 0) { m_s = -3.0e38f; s_s = 0.f; }
  __syncthreads();

  int beg = rowptr[node], end = rowptr[node + 1];

  // pass 1: scores + online softmax normalizers
  for (int j = beg; j < end; ++j) {
    int src = csrsrc[j];
    float e = edge_score<C>(xl, src, xr_s, att_s, red, tid);
    if (tid == 0) {
      int idx = j - beg;
      if (idx < SMAX) sc[idx] = e;
      float mo = m_s;
      float mn = fmaxf(mo, e);
      s_s = s_s * __expf(mo - mn) + __expf(e - mn);
      m_s = mn;
    }
    __syncthreads();
  }
  float m = m_s, sden = s_s;

  // pass 2: alpha-weighted accumulation of xl[src]
  float acc[C / AGG_T];
  #pragma unroll
  for (int q = 0; q < C / AGG_T; ++q) acc[q] = 0.f;

  for (int j = beg; j < end; ++j) {
    int src = csrsrc[j];
    int idx = j - beg;
    float e;
    if (idx < SMAX) e = sc[idx];
    else            e = edge_score<C>(xl, src, xr_s, att_s, red, tid); // rare overflow path
    float alpha = __expf(e - m) / sden;
    #pragma unroll
    for (int q = 0; q < C / AGG_T; ++q) {
      int c = tid + q * AGG_T;
      acc[q] += alpha * xl[(size_t)src * C + c];
    }
  }

  #pragma unroll
  for (int q = 0; q < C / AGG_T; ++q) {
    int c = tid + q * AGG_T;
    float o = acc[q] + bias[c];
    out[(size_t)node * C + c] = fmaxf(o, 0.f);   // fused network-level ReLU
  }
}

// ---------- global mean pool ----------
__global__ void k_pool(const float* __restrict__ h, const int* __restrict__ batch,
                       float* __restrict__ pooled, float* __restrict__ counts) {
  int idx = blockIdx.x * blockDim.x + threadIdx.x;
  int total = N_NODES * C2;
  if (idx >= total) return;
  int node = idx / C2;
  int c    = idx - node * C2;
  int g    = batch[node];
  atomicAdd(&pooled[g * C2 + c], h[idx]);
  if (c == 0) atomicAdd(&counts[g], 1.0f);
}

// ---------- classifier + sigmoid ----------
__global__ void k_classify(const float* __restrict__ pooled, const float* __restrict__ counts,
                           const float* __restrict__ Wc, const float* __restrict__ bc,
                           float* __restrict__ out) {
  int t = blockIdx.x * blockDim.x + threadIdx.x;
  if (t >= N_GRAPHS * N_CLASSES) return;
  int g = t >> 4, cls = t & 15;
  float inv = 1.0f / fmaxf(counts[g], 1.0f);
  float z = bc[cls];
  for (int c = 0; c < C2; ++c)
    z += pooled[g * C2 + c] * inv * Wc[c * N_CLASSES + cls];
  out[t] = 1.0f / (1.0f + __expf(-z));
}

// ---------------------------------------------------------------------------
extern "C" void kernel_launch(void* const* d_in, const int* in_sizes, int n_in,
                              void* d_out, int out_size, void* d_ws, size_t ws_size,
                              hipStream_t stream) {
  const float* x    = (const float*)d_in[0];
  const float* Wl1  = (const float*)d_in[1];
  const float* bl1  = (const float*)d_in[2];
  const float* Wr1  = (const float*)d_in[3];
  const float* br1  = (const float*)d_in[4];
  const float* att1 = (const float*)d_in[5];
  const float* b1   = (const float*)d_in[6];
  const float* Wl2  = (const float*)d_in[7];
  const float* bl2  = (const float*)d_in[8];
  const float* Wr2  = (const float*)d_in[9];
  const float* br2  = (const float*)d_in[10];
  const float* att2 = (const float*)d_in[11];
  const float* b2   = (const float*)d_in[12];
  const float* Wc   = (const float*)d_in[13];
  const float* bc   = (const float*)d_in[14];
  const int* ei     = (const int*)d_in[15];
  const int* batch  = (const int*)d_in[16];
  float* out        = (float*)d_out;

  char* ws = (char*)d_ws;
  size_t off = 0;
  auto carve = [&](size_t bytes) -> void* {
    void* p = ws + off;
    off = (off + bytes + 255) & ~(size_t)255;
    return p;
  };

  unsigned short* xbf   = (unsigned short*)carve((size_t)N_NODES * VOCAB * 2);
  unsigned short* Wl1p  = (unsigned short*)carve((size_t)VOCAB * C1 * 2);
  unsigned short* Wr1p  = (unsigned short*)carve((size_t)VOCAB * C1 * 2);
  float* xl1            = (float*)carve((size_t)N_NODES * C1 * 4);
  float* xr1            = (float*)carve((size_t)N_NODES * C1 * 4);
  float* h1             = (float*)carve((size_t)N_NODES * C1 * 4);
  unsigned short* h1bf  = (unsigned short*)carve((size_t)N_NODES * C1 * 2);
  unsigned short* Wl2p  = (unsigned short*)carve((size_t)C1 * C2 * 2);
  unsigned short* Wr2p  = (unsigned short*)carve((size_t)C1 * C2 * 2);
  float* xl2            = (float*)carve((size_t)N_NODES * C2 * 4);
  float* xr2            = (float*)carve((size_t)N_NODES * C2 * 4);
  float* h2             = (float*)carve((size_t)N_NODES * C2 * 4);
  int* deg              = (int*)carve((size_t)N_NODES * 4);
  int* rowptr           = (int*)carve((size_t)(N_NODES + 1) * 4);
  int* fill             = (int*)carve((size_t)N_NODES * 4);
  int* csrsrc           = (int*)carve((size_t)N_ETOT * 4);
  float* pooled         = (float*)carve((size_t)N_GRAPHS * C2 * 4);
  float* counts         = (float*)carve((size_t)N_GRAPHS * 4);

  // zero accumulators (graph-capturable)
  hipMemsetAsync(deg,    0, (size_t)N_NODES * 4, stream);
  hipMemsetAsync(fill,   0, (size_t)N_NODES * 4, stream);
  hipMemsetAsync(pooled, 0, (size_t)N_GRAPHS * C2 * 4, stream);
  hipMemsetAsync(counts, 0, (size_t)N_GRAPHS * 4, stream);

  auto blks = [](long long n, int t) { return (int)((n + t - 1) / t); };
  auto gemm_waves = [](int M, int N) -> long long {
    int tilesM = M >> 4, tilesN = N >> 4;
    return (long long)((tilesM + BM - 1) / BM) * (tilesN / BN);
  };

  // --- CSR build (shared by both layers) ---
  k_edge_count<<<blks(N_ETOT, 256), 256, 0, stream>>>(ei, deg);
  k_scan<<<1, 256, 0, stream>>>(deg, rowptr, N_NODES);
  k_edge_fill<<<blks(N_ETOT, 256), 256, 0, stream>>>(ei, rowptr, fill, csrsrc);

  // --- layer 1 GEMMs (bf16 WMMA) ---
  k_cvt_bf16<<<blks((long long)N_NODES * VOCAB, 256), 256, 0, stream>>>(x, xbf, N_NODES * VOCAB);
  k_pack_w<<<blks((long long)VOCAB * C1, 256), 256, 0, stream>>>(Wl1, Wl1p, VOCAB, C1);
  k_pack_w<<<blks((long long)VOCAB * C1, 256), 256, 0, stream>>>(Wr1, Wr1p, VOCAB, C1);
  {
    long long waves = gemm_waves(N_NODES, C1);
    k_gemm_bf16<<<blks(waves * 32, 256), 256, 0, stream>>>(xbf, Wl1p, bl1, xl1, N_NODES, C1, VOCAB);
    k_gemm_bf16<<<blks(waves * 32, 256), 256, 0, stream>>>(xbf, Wr1p, br1, xr1, N_NODES, C1, VOCAB);
  }

  // --- layer 1 attention aggregation (+bias+ReLU) ---
  k_gat_agg<C1><<<N_NODES, AGG_T, 0, stream>>>(xl1, xr1, att1, b1, rowptr, csrsrc, h1);

  // --- layer 2 GEMMs ---
  k_cvt_bf16<<<blks((long long)N_NODES * C1, 256), 256, 0, stream>>>(h1, h1bf, N_NODES * C1);
  k_pack_w<<<blks((long long)C1 * C2, 256), 256, 0, stream>>>(Wl2, Wl2p, C1, C2);
  k_pack_w<<<blks((long long)C1 * C2, 256), 256, 0, stream>>>(Wr2, Wr2p, C1, C2);
  {
    long long waves = gemm_waves(N_NODES, C2);
    k_gemm_bf16<<<blks(waves * 32, 256), 256, 0, stream>>>(h1bf, Wl2p, bl2, xl2, N_NODES, C2, C1);
    k_gemm_bf16<<<blks(waves * 32, 256), 256, 0, stream>>>(h1bf, Wr2p, br2, xr2, N_NODES, C2, C1);
  }

  // --- layer 2 attention aggregation (+bias+ReLU) ---
  k_gat_agg<C2><<<N_NODES, AGG_T, 0, stream>>>(xl2, xr2, att2, b2, rowptr, csrsrc, h2);

  // --- mean pool + classifier ---
  k_pool<<<blks((long long)N_NODES * C2, 256), 256, 0, stream>>>(h2, batch, pooled, counts);
  k_classify<<<blks(N_GRAPHS * N_CLASSES, 256), 256, 0, stream>>>(pooled, counts, Wc, bc, out);
}
